// LoRALinear_67774583931082
// MI455X (gfx1250) — compile-verified, matrix operand order checked by reading, not run on hardware
//
#include <hip/hip_runtime.h>

#define BATCH 8
#define SEQ   2048
#define DIN   2048
#define DOUT  2048
#define RANK  16

#define BM 128
#define BN 128
#define BK 32
#define LDS_STRIDE (BK + 8)   // 40 bf16 = 80 B row pitch (TDM pad: 16 DW interval + 4 DW pad)

typedef __bf16 bf16_t;
typedef bf16_t v4bf  __attribute__((ext_vector_type(4)));
typedef bf16_t v8bf  __attribute__((ext_vector_type(8)));
typedef bf16_t v16bf __attribute__((ext_vector_type(16)));
typedef float  v8f   __attribute__((ext_vector_type(8)));
typedef unsigned int v4u __attribute__((ext_vector_type(4)));
typedef int  v4i __attribute__((ext_vector_type(4)));
typedef int  v8i __attribute__((ext_vector_type(8)));

union FragU { v16bf v; v8bf h[2]; };

__device__ __forceinline__ bf16_t f32_to_bf16(float f) {
  unsigned u = __builtin_bit_cast(unsigned, f);
  u += 0x7FFFu + ((u >> 16) & 1u);           // round-to-nearest-even
  unsigned short h = (unsigned short)(u >> 16);
  return __builtin_bit_cast(bf16_t, h);
}

// ---------------------------------------------------------------------------
// TDM tile load: 2D tile of bf16 (tile_d0 x tile_d1 elements), row stride
// stride0 elements in memory, into LDS with 16B padding after every 64B row
// (row pitch 80B == LDS_STRIDE). OOB columns (>= tensor_d0) are zero-filled.
// Descriptor bitfields per CDNA5 ISA §8.3/8.4. 6-arg builtin (clang-23 form).
// ---------------------------------------------------------------------------
__device__ __forceinline__ void tdm_load_tile(unsigned lds_addr,
                                              unsigned long long gaddr,
                                              unsigned tensor_d0, unsigned stride0,
                                              unsigned tile_d0,  unsigned tile_d1,
                                              unsigned tensor_d1) {
  v4u g0;
  g0[0] = 1u;                                              // count=1 (valid), user mode
  g0[1] = lds_addr;                                        // LDS byte address
  g0[2] = (unsigned)(gaddr & 0xFFFFFFFFull);               // global_addr[31:0]
  g0[3] = (unsigned)((gaddr >> 32) & 0x1FFFFFFull)         // global_addr[56:32]
        | (2u << 30);                                      // type = 2 ("image")
  v8i g1;
  g1[0] = (int)((1u << 16)        // data_size = 1 (2 bytes)
              | (1u << 20)        // pad_enable
              | (3u << 22)        // pad_interval = 16 DWORDs (64 B)
              | (3u << 25));      // pad_amount   = 4 DWORDs (16 B)
  g1[1] = (int)((tensor_d0 & 0xFFFFu) << 16);              // tensor_dim0[15:0]
  g1[2] = (int)(((tensor_d0 >> 16) & 0xFFFFu)              // tensor_dim0[31:16]
              | ((tensor_d1 & 0xFFFFu) << 16));            // tensor_dim1[15:0]
  g1[3] = (int)(((tensor_d1 >> 16) & 0xFFFFu)              // tensor_dim1[31:16]
              | ((tile_d0 & 0xFFFFu) << 16));              // tile_dim0
  g1[4] = (int)(tile_d1 & 0xFFFFu);                        // tile_dim1 (tile_dim2 = 0)
  g1[5] = (int)stride0;                                    // tensor_dim0_stride[31:0]
  g1[6] = 0;                                               // stride[47:32] | dim1_stride lo
  g1[7] = 0;
  v4i z4 = {0, 0, 0, 0};
  v8i z8 = {0, 0, 0, 0, 0, 0, 0, 0};
  __builtin_amdgcn_tensor_load_to_lds(g0, g1, z4, z4, z8, 0);
}

// ---------------------------------------------------------------------------
// Kernel 1: T[b,s,r] = sum_i x[b,s,i] * B_params[b,r,i]   (bf16 out)
// ---------------------------------------------------------------------------
__global__ __launch_bounds__(256)
void lora_intermediate_kernel(const float* __restrict__ x,
                              const float* __restrict__ Bp,
                              bf16_t* __restrict__ T) {
  const int tid  = threadIdx.x;
  const int r    = tid & (RANK - 1);
  const int si   = tid >> 4;
  const int sblk = blockIdx.x % (SEQ / 16);
  const int b    = blockIdx.x / (SEQ / 16);
  const int s    = sblk * 16 + si;

  const float4* xr = (const float4*)(x  + ((size_t)b * SEQ  + s) * DIN);
  const float4* br = (const float4*)(Bp + ((size_t)b * RANK + r) * DIN);

  float acc = 0.f;
#pragma unroll 4
  for (int i = 0; i < DIN / 4; ++i) {
    float4 a = xr[i];
    float4 w = br[i];
    acc += a.x * w.x + a.y * w.y + a.z * w.z + a.w * w.w;
  }
  T[((size_t)b * SEQ + s) * RANK + r] = f32_to_bf16(acc);
}

// ---------------------------------------------------------------------------
// Kernel 2: vectorized fp32 -> bf16 (x, W, A_params pre-conversion)
// ---------------------------------------------------------------------------
__global__ __launch_bounds__(256)
void convert_bf16x4_kernel(const float* __restrict__ src, bf16_t* __restrict__ dst, int n4) {
  int i = blockIdx.x * 256 + threadIdx.x;
  if (i < n4) {
    float4 v = ((const float4*)src)[i];
    v4bf p;
    p[0] = f32_to_bf16(v.x); p[1] = f32_to_bf16(v.y);
    p[2] = f32_to_bf16(v.z); p[3] = f32_to_bf16(v.w);
    ((v4bf*)dst)[i] = p;
  }
}

// ---------------------------------------------------------------------------
// Kernel 3: out = x@W^T + bias + T@A^T, all-bf16 WMMA, TDM-fed LDS tiles.
// 65 K-steps: 64 of the base GEMM + 1 zero-padded rank-16 LoRA step
// (TDM OOB zero-fill handles K=16..31). Double-buffered: wave 0 issues the
// next tile pair while all waves compute on the current one.
// ---------------------------------------------------------------------------
__global__ __launch_bounds__(256)
void lora_gemm_tdm_kernel(const bf16_t* __restrict__ xb,
                          const bf16_t* __restrict__ wb,
                          const float*  __restrict__ bias,
                          const bf16_t* __restrict__ T,
                          const bf16_t* __restrict__ Abf,
                          float* __restrict__ out) {
  __shared__ __align__(16) bf16_t sA[2][BM * LDS_STRIDE];
  __shared__ __align__(16) bf16_t sB[2][BN * LDS_STRIDE];

  const int tid = threadIdx.x;
  const int n0  = blockIdx.x * BN;
  const int m0  = blockIdx.y * BM;
  const int b   = blockIdx.z;

  const int wave = tid >> 5;
  const int lane = tid & 31;
  const int wm   = (wave >> 2) * 64;   // wave M offset
  const int wn   = (wave & 3) * 32;    // wave N offset
  const int lr   = lane & 15;
  const int ko   = (lane >> 4) * 8;

  const unsigned ldsA0 = (unsigned)(uintptr_t)&sA[0][0];
  const unsigned ldsA1 = (unsigned)(uintptr_t)&sA[1][0];
  const unsigned ldsB0 = (unsigned)(uintptr_t)&sB[0][0];
  const unsigned ldsB1 = (unsigned)(uintptr_t)&sB[1][0];

  const unsigned long long gA  = (unsigned long long)(uintptr_t)(xb + ((size_t)b * SEQ + m0) * DIN);
  const unsigned long long gB  = (unsigned long long)(uintptr_t)(wb + (size_t)n0 * DIN);
  const unsigned long long gT  = (unsigned long long)(uintptr_t)(T  + ((size_t)b * SEQ  + m0) * RANK);
  const unsigned long long gAb = (unsigned long long)(uintptr_t)(Abf + ((size_t)b * DOUT + n0) * RANK);

  const int NSTEPS = DIN / BK + 1;     // 64 base + 1 LoRA step

  auto issue = [&](int step, int buf) {
    const unsigned la = buf ? ldsA1 : ldsA0;
    const unsigned lb = buf ? ldsB1 : ldsB0;
    if (step < DIN / BK) {
      const unsigned long long off = (unsigned long long)step * BK * sizeof(bf16_t);
      tdm_load_tile(la, gA + off, BK, DIN, BK, BM, BM);    // x tile  128x32
      tdm_load_tile(lb, gB + off, BK, DIN, BK, BN, BN);    // W tile  128x32
    } else {
      // LoRA step: 16 real K + 16 zero-filled (tensor_d0=16 < tile_d0=32)
      tdm_load_tile(la, gT,  RANK, RANK, BK, BM, BM);      // T tile
      tdm_load_tile(lb, gAb, RANK, RANK, BK, BN, BN);      // A tile
    }
  };

  v8f acc[4][2];
#pragma unroll
  for (int mi = 0; mi < 4; ++mi)
#pragma unroll
    for (int ni = 0; ni < 2; ++ni)
#pragma unroll
      for (int j = 0; j < 8; ++j) acc[mi][ni][j] = 0.f;

  if (wave == 0) issue(0, 0);

  for (int i = 0; i < NSTEPS; ++i) {
    const int cur = i & 1;
    if (wave == 0) {
      if (i + 1 < NSTEPS) {
        issue(i + 1, cur ^ 1);                   // overlap DMA of next tile
        __builtin_amdgcn_s_wait_tensorcnt(2);    // current tile pair complete
      } else {
        __builtin_amdgcn_s_wait_tensorcnt(0);
      }
    }
    __syncthreads();                             // tile ready for all waves

    v16bf afrag[4], bfrag[2];
#pragma unroll
    for (int mi = 0; mi < 4; ++mi) {
      const bf16_t* p = &sA[cur][(wm + mi * 16 + lr) * LDS_STRIDE + ko];
      FragU f;
      f.h[0] = *(const v8bf*)p;                  // K = ko .. ko+7
      f.h[1] = *(const v8bf*)(p + 16);           // K = ko+16 .. ko+23
      afrag[mi] = f.v;
    }
#pragma unroll
    for (int ni = 0; ni < 2; ++ni) {
      const bf16_t* p = &sB[cur][(wn + ni * 16 + lr) * LDS_STRIDE + ko];
      FragU f;
      f.h[0] = *(const v8bf*)p;
      f.h[1] = *(const v8bf*)(p + 16);
      bfrag[ni] = f.v;
    }
#pragma unroll
    for (int mi = 0; mi < 4; ++mi)
#pragma unroll
      for (int ni = 0; ni < 2; ++ni)
        acc[mi][ni] = __builtin_amdgcn_wmma_f32_16x16x32_bf16(
            false, afrag[mi], false, bfrag[ni],
            (short)0, acc[mi][ni], false, false);

    __syncthreads();                             // all done reading before overwrite
  }

  // ---- epilogue: bias add + store (C/D layout: VGPR j -> M = j + (lane>>4)*8)
#pragma unroll
  for (int ni = 0; ni < 2; ++ni) {
    const int col = n0 + wn + ni * 16 + lr;
    const float bv = bias[col];
#pragma unroll
    for (int mi = 0; mi < 4; ++mi) {
#pragma unroll
      for (int j = 0; j < 8; ++j) {
        const int mrow = m0 + wm + mi * 16 + ko + j;   // ko == (lane>>4)*8
        out[((size_t)b * SEQ + mrow) * DOUT + col] = acc[mi][ni][j] + bv;
      }
    }
  }
}

// ---------------------------------------------------------------------------
extern "C" void kernel_launch(void* const* d_in, const int* in_sizes, int n_in,
                              void* d_out, int out_size, void* d_ws, size_t ws_size,
                              hipStream_t stream) {
  const float* x    = (const float*)d_in[0];  // [B, S, DIN]
  const float* Ap   = (const float*)d_in[1];  // [B, DOUT, R]
  const float* Bp   = (const float*)d_in[2];  // [B, R, DIN]
  const float* W    = (const float*)d_in[3];  // [DOUT, DIN]
  const float* bias = (const float*)d_in[4];  // [DOUT]
  float* out = (float*)d_out;                 // [B, S, DOUT]

  // workspace: bf16 copies (x, W), LoRA intermediate T, bf16 A
  bf16_t* xb  = (bf16_t*)d_ws;                          // B*S*DIN
  bf16_t* wbf = xb  + (size_t)BATCH * SEQ * DIN;        // DOUT*DIN
  bf16_t* T   = wbf + (size_t)DOUT * DIN;               // B*S*R
  bf16_t* Abf = T   + (size_t)BATCH * SEQ * RANK;       // B*DOUT*R

  const int nX4 = BATCH * SEQ * DIN / 4;
  const int nW4 = DOUT * DIN / 4;
  const int nA4 = BATCH * DOUT * RANK / 4;

  convert_bf16x4_kernel<<<dim3((nX4 + 255) / 256), dim3(256), 0, stream>>>(x, xb, nX4);
  convert_bf16x4_kernel<<<dim3((nW4 + 255) / 256), dim3(256), 0, stream>>>(W, wbf, nW4);
  convert_bf16x4_kernel<<<dim3((nA4 + 255) / 256), dim3(256), 0, stream>>>(Ap, Abf, nA4);
  lora_intermediate_kernel<<<dim3(BATCH * (SEQ / 16)), dim3(256), 0, stream>>>(x, Bp, T);

  lora_gemm_tdm_kernel<<<dim3(DOUT / BN, SEQ / BM, BATCH), dim3(256), 0, stream>>>(
      xb, wbf, bias, T, Abf, out);
}
